// QuantLinear_39427799777830
// MI455X (gfx1250) — matrix operand order repaired and checked by reading, not compiled, not run on hardware
//
#include <hip/hip_runtime.h>

// GPTQ 4-bit dequant + GEMM via v_wmma_f32_16x16x32_f16 on gfx1250 (wave32).
// out[M,N] = x[M,K] @ W[K,N] + bias,  W = scale[n] * (nib(k,n) - (zero[n]+1)).
// Single quant group (g_idx == 0 everywhere, per the reference setup).
//
// Dequant uses the f16 magic-number trick: (nib | 0x6400) is f16(1024+nib)
// exactly; v_pk_add_f16 subtracts the exact integer (1024+zero+1), then
// v_pk_mul_f16 applies the scale. This yields K micro-order (0,4,1,5,2,6,3,7)
// per 8-group; the x tile is staged with the same permutation, so the WMMA
// dot product is unchanged.

typedef __attribute__((ext_vector_type(2)))  _Float16 v2h;
typedef __attribute__((ext_vector_type(8)))  _Float16 v8h;
typedef __attribute__((ext_vector_type(16))) _Float16 v16h;
typedef __attribute__((ext_vector_type(8)))  float    v8f;

#define BM 128
#define BN 128
#define BK 64
#define LDA 72   // BK + 8 halfs of padding (16B) -> conflict-free 16-lane frag loads
#define LDB 72
#define THREADS 256

__device__ __forceinline__ v16h join16(v8h lo, v8h hi) {
  v16h r;
#pragma unroll
  for (int i = 0; i < 8; ++i) { r[i] = lo[i]; r[i + 8] = hi[i]; }
  return r;
}

__global__ __launch_bounds__(THREADS)
void gptq_gemm_wmma(const float* __restrict__ x,
                    const int*   __restrict__ qweight,
                    const int*   __restrict__ qzeros,
                    const float* __restrict__ scales,
                    const float* __restrict__ bias,
                    float* __restrict__ out,
                    int M, int N, int K)
{
  __shared__ _Float16 As[2][BM * LDA];  // x tile, f16, [BM][BK], permuted K micro-order
  __shared__ _Float16 Ws[2][BN * LDB];  // weight tile, f16, [BN][BK], same micro-order

  const int t    = threadIdx.x;
  const int lane = t & 31;
  const int wave = t >> 5;       // 8 waves
  const int wm   = wave & 3;     // 4 waves along M
  const int wn   = wave >> 2;    // 2 waves along N
  const int l16  = lane & 15;
  const int lhi  = lane >> 4;    // 0 or 1

  const int m0 = blockIdx.y * BM;
  const int n0 = blockIdx.x * BN;

  // ---- per-thread dequant column params (thread's weight column is fixed) ----
  const int      wc   = t & 127;            // column within tile (0..127)
  const int      wr0  = t >> 7;             // first packed row (0 or 1), step 2
  const int      ncol = n0 + wc;
  const float    s    = scales[ncol];
  const int      zq   = (qzeros[ncol >> 3] >> ((ncol & 7) * 4)) & 15;
  const _Float16 sh   = (_Float16)s;
  const _Float16 bh   = (_Float16)(-(float)(zq + 1 + 1024));  // exact integer in f16
  const v2h      s2   = { sh, sh };
  const v2h      b2   = { bh, bh };

  // ---- A copy indices: thread owns one 8-float K chunk per row ----
  const int ac8 = t & 7;                // K chunk (0..7): K = ac8*8 .. +7
  const int ar0 = t >> 3;               // first row (0..31), step 32

  const int numT = K / BK;

  v8f acc[2][4];
#pragma unroll
  for (int i = 0; i < 2; ++i)
#pragma unroll
    for (int j = 0; j < 4; ++j) acc[i][j] = (v8f)0.0f;

  float4 pfx[8];
  int    pfq[4];

  auto prefetch = [&](int kt) {
#pragma unroll
    for (int i = 0; i < 4; ++i) {
      const float* p = &x[(size_t)(m0 + ar0 + 32 * i) * K + kt + ac8 * 8];
      pfx[2 * i]     = *(const float4*)p;
      pfx[2 * i + 1] = *(const float4*)(p + 4);
    }
#pragma unroll
    for (int i = 0; i < 4; ++i)
      pfq[i] = qweight[(size_t)((kt >> 3) + wr0 + 2 * i) * N + ncol];
    // hint the streams one more BK-tile ahead (global_prefetch_b8)
    if (kt + BK < K) {
      __builtin_prefetch(&x[(size_t)(m0 + ar0) * K + kt + BK + ac8 * 8], 0, 0);
      __builtin_prefetch(&qweight[(size_t)(((kt + BK) >> 3) + wr0) * N + ncol], 0, 0);
    }
  };

  auto stage = [&](int b) {
#pragma unroll
    for (int i = 0; i < 4; ++i) {
      const float4 lo = pfx[2 * i], hi = pfx[2 * i + 1];
      // permuted K micro-order: positions (0..7) <- K (0,4,1,5,2,6,3,7)
      v8h h = { (_Float16)lo.x, (_Float16)hi.x,
                (_Float16)lo.y, (_Float16)hi.y,
                (_Float16)lo.z, (_Float16)hi.z,
                (_Float16)lo.w, (_Float16)hi.w };
      *(v8h*)&As[b][(ar0 + 32 * i) * LDA + ac8 * 8] = h;
    }
#pragma unroll
    for (int i = 0; i < 4; ++i) {
      const unsigned q = (unsigned)pfq[i];
      v8h h;
#pragma unroll
      for (int j = 0; j < 4; ++j) {
        // nibbles j (bits 4j..) and j+4 (bits 4j+16..) -> two f16 lanes
        unsigned u = ((q >> (4 * j)) & 0x000F000Fu) | 0x64006400u;
        v2h hv = __builtin_bit_cast(v2h, u);
        v2h w  = (hv + b2) * s2;              // v_pk_add_f16 + v_pk_mul_f16
        h[2 * j]     = w[0];                   // K = k0 + j
        h[2 * j + 1] = w[1];                   // K = k0 + j + 4
      }
      *(v8h*)&Ws[b][wc * LDB + (wr0 + 2 * i) * 8] = h;
    }
  };

  auto compute = [&](int b) {
#pragma unroll
    for (int ks = 0; ks < 2; ++ks) {            // two K=32 WMMA steps per BK=64
      const int kA = ks * 32 + lhi * 8;         // A lane layout: 8-half groups
      const int kB = ks * 32 + lhi * 8;         // B lane layout: 8-half groups
      v16h afrag[2], bfrag[4];
#pragma unroll
      for (int fm = 0; fm < 2; ++fm) {
        const _Float16* p = &As[b][(wm * 32 + fm * 16 + l16) * LDA + kA];
        afrag[fm] = join16(*(const v8h*)p, *(const v8h*)(p + 16));
      }
#pragma unroll
      for (int fn = 0; fn < 4; ++fn) {
        const _Float16* p = &Ws[b][(wn * 64 + fn * 16 + l16) * LDB + kB];
        bfrag[fn] = join16(*(const v8h*)p, *(const v8h*)(p + 16));
      }
#pragma unroll
      for (int fm = 0; fm < 2; ++fm)
#pragma unroll
        for (int fn = 0; fn < 4; ++fn)
          acc[fm][fn] = __builtin_amdgcn_wmma_f32_16x16x32_f16(
              false, afrag[fm], false, bfrag[fn],
              (short)0, acc[fm][fn], false, false);
    }
  };

  // ---- pipelined main loop: one barrier per tile, LDS double buffered ----
  prefetch(0);
  stage(0);
  __syncthreads();

  for (int tt = 0; tt < numT; ++tt) {
    const int cur = tt & 1;
    if (tt + 1 < numT) prefetch((tt + 1) * BK);
    compute(cur);
    if (tt + 1 < numT) stage(cur ^ 1);
    __syncthreads();
  }

  // ---- epilogue: C/D layout -> global (VGPR r = row r / r+8, lane = col) ----
#pragma unroll
  for (int fn = 0; fn < 4; ++fn) {
    const int col = n0 + wn * 64 + fn * 16 + l16;
    const float bv = bias[col];
#pragma unroll
    for (int fm = 0; fm < 2; ++fm) {
      const int rbase = m0 + wm * 32 + fm * 16 + lhi * 8;
#pragma unroll
      for (int r = 0; r < 8; ++r)
        out[(size_t)(rbase + r) * N + col] = acc[fm][fn][r] + bv;
    }
  }
}

extern "C" void kernel_launch(void* const* d_in, const int* in_sizes, int n_in,
                              void* d_out, int out_size, void* d_ws, size_t ws_size,
                              hipStream_t stream) {
  const float* x       = (const float*)d_in[0];
  const int*   qweight = (const int*)  d_in[1];
  const int*   qzeros  = (const int*)  d_in[2];
  const float* scales  = (const float*)d_in[3];
  // d_in[4] = g_idx (all zeros; single group -> unused)
  const float* bias    = (const float*)d_in[5];
  float* out = (float*)d_out;

  const int K = in_sizes[4];          // infeatures  (g_idx length)
  const int N = in_sizes[5];          // outfeatures (bias length)
  const int M = in_sizes[0] / K;      // tokens

  dim3 grid(N / BN, M / BM);
  dim3 block(THREADS);
  gptq_gemm_wmma<<<grid, block, 0, stream>>>(x, qweight, qzeros, scales, bias,
                                             out, M, N, K);
}
